// AttentionGCNLayer_20177756357085
// MI455X (gfx1250) — matrix operand order repaired and checked by reading, not compiled, not run on hardware
//
#include <hip/hip_runtime.h>
#include <hip/hip_fp16.h>
#include <math.h>

typedef __attribute__((ext_vector_type(16))) _Float16 v16h;
typedef __attribute__((ext_vector_type(8)))  float    v8f;

#define N_NODES 4096
#define IN_DIM  256
#define OUT_DIM 256
#define HEADS   4
#define DK      64
#define NEG_SLOPE 0.2f
#define LOG2E 1.4426950408889634f

__device__ __forceinline__ float lrelu(float x) { return x >= 0.f ? x : NEG_SLOPE * x; }
// raw v_exp_f32 (base-2) — inputs are pre-scaled by log2(e)
__device__ __forceinline__ float fast_exp2(float x) { return __builtin_amdgcn_exp2f(x); }

struct alignas(16) Half8 { _Float16 h[8]; };
union HU { Half8 q[2]; v16h v; };

// ---------------------------------------------------------------------------
// Kernel 1: Wh = H @ W^T  (f16 WMMA, f32 accumulate)
// One wave per 16x16 output tile. Also emits WhT16 (transposed f16 copy) for
// the aggregation GEMM's B operand.
// ---------------------------------------------------------------------------
__global__ void __launch_bounds__(32)
gemm_wh_kernel(const float* __restrict__ H, const float* __restrict__ W,
               float* __restrict__ Wh, _Float16* __restrict__ WhT16) {
    int lane = threadIdx.x;            // 0..31
    int Mb = blockIdx.x * 16;          // node tile
    int Ob = blockIdx.y * 16;          // output-feature tile
    int l15 = lane & 15;
    int hi  = lane >> 4;               // lane half
    v8f c = {};
    const float* hrow = H + (size_t)(Mb + l15) * IN_DIM;
    const float* wrow = W + (size_t)(Ob + l15) * IN_DIM;
    for (int kb = 0; kb < IN_DIM; kb += 32) {
        // A fragment (16x32 f16): lanes 0-15 hold K {0..15}, lanes 16-31 {16..31}
        int k0 = kb + hi * 16;
        v16h a;
        #pragma unroll
        for (int t = 0; t < 8; ++t) {
            a[t]     = (_Float16)hrow[k0 + t];
            a[8 + t] = (_Float16)hrow[k0 + 8 + t];
        }
        // B fragment (32x16 f16): lane holds column N=l15, K = hi*16 .. +15 (contiguous)
        int kw = kb + hi * 16;
        v16h b;
        #pragma unroll
        for (int t = 0; t < 16; ++t) b[t] = (_Float16)wrow[kw + t];
        c = __builtin_amdgcn_wmma_f32_16x16x32_f16(false, a, false, b, (short)0, c, false, false);
    }
    // C layout: VGPR k -> row = k + 8*hi, col = l15
    int ocol  = Ob + l15;
    int rbase = Mb + hi * 8;
    Half8 t16;
    #pragma unroll
    for (int k = 0; k < 8; ++k) {
        float v = c[k];
        Wh[(size_t)(rbase + k) * OUT_DIM + ocol] = v;
        t16.h[k] = (_Float16)v;
    }
    // transposed f16: row = feature, contiguous in node index -> single 16B store
    *reinterpret_cast<Half8*>(&WhT16[(size_t)ocol * N_NODES + rbase]) = t16;
}

// ---------------------------------------------------------------------------
// Kernel 2: attention scores, head-major layout, pre-scaled by log2(e) so the
// hot loops can use the native v_exp_f32 (exp2) path without a multiply.
// LeakyReLU commutes with positive scaling, so softmax values are unchanged.
// ---------------------------------------------------------------------------
__global__ void __launch_bounds__(256)
score_kernel(const float* __restrict__ Wh, const float* __restrict__ a_l,
             const float* __restrict__ a_r, float* __restrict__ slT,
             float* __restrict__ srT) {
    int idx = blockIdx.x * blockDim.x + threadIdx.x;   // 0..16383
    int n = idx >> 2, h = idx & 3;
    const float* w  = Wh + (size_t)n * OUT_DIM + h * DK;
    const float* al = a_l + h * DK;
    const float* ar = a_r + h * DK;
    float sl = 0.f, sr = 0.f;
    #pragma unroll 8
    for (int d = 0; d < DK; ++d) { float v = w[d]; sl += v * al[d]; sr += v * ar[d]; }
    slT[h * N_NODES + n] = sl * LOG2E;
    srT[h * N_NODES + n] = sr * LOG2E;
}

// ---------------------------------------------------------------------------
// Kernel 3: per-row stats. Packs mask bits (A>0 || self-loop), computes exact
// softmax row max (LeakyReLU monotone => max_j LR(sl+sr_j) = LR(sl+max sr_j))
// and partition sum Z (all in the log2-scaled domain). One block per row.
// ---------------------------------------------------------------------------
__global__ void __launch_bounds__(256)
rowstats_kernel(const float* __restrict__ A, const float* __restrict__ slT,
                const float* __restrict__ srT, unsigned* __restrict__ maskbits,
                float* __restrict__ emaxArr, float* __restrict__ zArr) {
    __shared__ unsigned mw[N_NODES / 32];
    __shared__ float red[256];
    int i = blockIdx.x;
    int tid = threadIdx.x;
    if (tid < 128) {
        unsigned wbits = 0;
        const float* arow = A + (size_t)i * N_NODES + tid * 32;
        #pragma unroll
        for (int b = 0; b < 32; ++b) {
            int j = tid * 32 + b;
            if (arow[b] > 0.f || j == i) wbits |= (1u << b);
        }
        mw[tid] = wbits;
        maskbits[(size_t)i * 128 + tid] = wbits;
    }
    __syncthreads();
    for (int h = 0; h < HEADS; ++h) {
        const float* sr = srT + h * N_NODES;
        float m = -INFINITY;
        for (int j = tid; j < N_NODES; j += 256)
            if ((mw[j >> 5] >> (j & 31)) & 1) m = fmaxf(m, sr[j]);
        red[tid] = m; __syncthreads();
        #pragma unroll
        for (int s = 128; s > 0; s >>= 1) {
            if (tid < s) red[tid] = fmaxf(red[tid], red[tid + s]);
            __syncthreads();
        }
        float sl = slT[h * N_NODES + i];
        float em = lrelu(sl + red[0]);
        __syncthreads();
        float z = 0.f;
        for (int j = tid; j < N_NODES; j += 256)
            if ((mw[j >> 5] >> (j & 31)) & 1) z += fast_exp2(lrelu(sl + sr[j]) - em);
        red[tid] = z; __syncthreads();
        #pragma unroll
        for (int s = 128; s > 0; s >>= 1) {
            if (tid < s) red[tid] += red[tid + s];
            __syncthreads();
        }
        if (tid == 0) { emaxArr[h * N_NODES + i] = em; zArr[h * N_NODES + i] = red[0]; }
        __syncthreads();
    }
}

// ---------------------------------------------------------------------------
// Kernel 4: out[i, h*64+d] = elu( (sum_j P~[i,j] * Wh[j, h*64+d]) / Z_i )
// Block = (16-row tile, head), 4 waves split the neighbor (K) dimension so
// each exp2 is computed exactly once. WMMA f16 with f32 accumulate; LDS
// cross-wave reduction + normalize + ELU in the epilogue. Prefetches the
// next K-chunk of the B operand (global_prefetch_b8) to overlap L2 latency
// with the exp2/WMMA work.
// ---------------------------------------------------------------------------
__global__ void __launch_bounds__(128)
agg_kernel(const unsigned* __restrict__ maskbits, const float* __restrict__ slT,
           const float* __restrict__ srT, const float* __restrict__ emaxArr,
           const float* __restrict__ zArr, const _Float16* __restrict__ WhT16,
           float* __restrict__ out) {
    __shared__ float sp[4 * 4 * 256];      // [wave][ntile][k*32+lane], 16KB
    int tid  = threadIdx.x;
    int wave = tid >> 5, lane = tid & 31;
    int l15  = lane & 15, hi = lane >> 4;
    int i0 = blockIdx.x * 16;
    int h  = blockIdx.y;
    int irow = i0 + l15;                   // A-fragment row for this lane
    float sl = slT[h * N_NODES + irow];
    float em = emaxArr[h * N_NODES + irow];
    v8f cacc[4] = {v8f{}, v8f{}, v8f{}, v8f{}};
    int jstart = wave * (N_NODES / 4);
    for (int kb = 0; kb < N_NODES / 4; kb += 32) {
        int jb = jstart + kb;
        unsigned mword = maskbits[(size_t)irow * 128 + (jb >> 5)];
        const float* sr = srT + h * N_NODES + jb;
        int o1 = hi * 16;      // lane-half K offsets within the 32-block
        int o2 = o1 + 8;
        // Build P~ A-fragment in f16 (values in [0,1]); exp2 in scaled domain
        v16h a;
        #pragma unroll
        for (int t = 0; t < 8; ++t) {
            float p1 = ((mword >> (o1 + t)) & 1) ? fast_exp2(lrelu(sl + sr[o1 + t]) - em) : 0.f;
            float p2 = ((mword >> (o2 + t)) & 1) ? fast_exp2(lrelu(sl + sr[o2 + t]) - em) : 0.f;
            a[t]     = (_Float16)p1;
            a[8 + t] = (_Float16)p2;
        }
        int jcol = jb + hi * 16;
        #pragma unroll
        for (int nt = 0; nt < 4; ++nt) {
            const _Float16* bp = WhT16 + (size_t)(h * DK + nt * 16 + l15) * N_NODES + jcol;
            HU bu;
            bu.q[0] = *reinterpret_cast<const Half8*>(bp);
            bu.q[1] = *reinterpret_cast<const Half8*>(bp + 8);
            // prefetch next K-chunk of this B row (predictable +64B stride)
            __builtin_prefetch(bp + 32, 0, 1);
            cacc[nt] = __builtin_amdgcn_wmma_f32_16x16x32_f16(false, a, false, bu.v,
                                                              (short)0, cacc[nt], false, false);
        }
        __builtin_prefetch(sr + 32, 0, 1);
    }
    // Cross-wave reduction in LDS
    #pragma unroll
    for (int nt = 0; nt < 4; ++nt)
        #pragma unroll
        for (int k = 0; k < 8; ++k)
            sp[(wave * 4 + nt) * 256 + k * 32 + lane] = cacc[nt][k];
    __syncthreads();
    {
        int nt = wave;   // each wave finalizes one 16-col tile
        #pragma unroll
        for (int k = 0; k < 8; ++k) {
            int idx = nt * 256 + k * 32 + lane;
            float v = sp[idx] + sp[4 * 256 + idx] + sp[8 * 256 + idx] + sp[12 * 256 + idx];
            int row = k + hi * 8;
            int gi  = i0 + row;
            float z = zArr[h * N_NODES + gi];
            float o = v / z;
            o = o > 0.f ? o : (__expf(o) - 1.f);   // ELU
            out[(size_t)gi * OUT_DIM + h * DK + nt * 16 + l15] = o;
        }
    }
}

// ---------------------------------------------------------------------------
extern "C" void kernel_launch(void* const* d_in, const int* in_sizes, int n_in,
                              void* d_out, int out_size, void* d_ws, size_t ws_size,
                              hipStream_t stream) {
    const float* H   = (const float*)d_in[0];
    const float* A   = (const float*)d_in[1];
    const float* W   = (const float*)d_in[2];
    const float* a_l = (const float*)d_in[3];
    const float* a_r = (const float*)d_in[4];
    float* out = (float*)d_out;

    char* ws = (char*)d_ws;
    float*    Wh       = (float*)(ws);                               // 4 MB
    _Float16* WhT16    = (_Float16*)(ws + 4u * 1024 * 1024);         // 2 MB
    float*    slT      = (float*)(ws + 6u * 1024 * 1024);            // 64 KB
    float*    srT      = (float*)(ws + 6u * 1024 * 1024 + 65536);    // 64 KB
    float*    emaxArr  = (float*)(ws + 6u * 1024 * 1024 + 2 * 65536);
    float*    zArr     = (float*)(ws + 6u * 1024 * 1024 + 3 * 65536);
    unsigned* maskbits = (unsigned*)(ws + 6u * 1024 * 1024 + 4 * 65536); // 2 MB

    gemm_wh_kernel<<<dim3(N_NODES / 16, OUT_DIM / 16), 32, 0, stream>>>(H, W, Wh, WhT16);
    score_kernel<<<(N_NODES * HEADS) / 256, 256, 0, stream>>>(Wh, a_l, a_r, slT, srT);
    rowstats_kernel<<<N_NODES, 256, 0, stream>>>(A, slT, srT, maskbits, emaxArr, zArr);
    agg_kernel<<<dim3(N_NODES / 16, HEADS), 128, 0, stream>>>(maskbits, slT, srT,
                                                              emaxArr, zArr, WhT16, out);
}
